// BiRWKV7Block_78391743086641
// MI455X (gfx1250) — matrix operand order repaired and checked by reading, not compiled, not run on hardware
//
#include <hip/hip_runtime.h>
#include <hip/hip_bf16.h>

// ---------------------------------------------------------------------------
// BiRWKV7 block for MI455X (gfx1250, wave32, WMMA).
//
// Design (see analysis): all dense projections run as bf16 WMMA GEMMs
// (v_wmma_f32_16x16x32_bf16) with f32 accumulation; the token-shift mix
// (xn + diff*coef) is fused into the A-tile staging so the six mixed input
// tensors are never materialized. Weights are converted once to transposed
// bf16 ([N,K]) so A and B fragments are contiguous 16B LDS reads.
// The RWKV-7 scan runs one 64-thread block per (dir,b,h) with the 64x64 f32
// state resident in VGPRs (thread i owns state row i).
// ---------------------------------------------------------------------------

typedef __bf16 bf16_t;
typedef __attribute__((ext_vector_type(16))) __bf16 v16bf;
typedef __attribute__((ext_vector_type(8)))  __bf16 v8bf;
typedef __attribute__((ext_vector_type(8)))  float  v8f;

namespace {
constexpr int Bb = 4, Tt = 2048, Cc = 512, Nn = 64, FF = 2048;
constexpr int M1 = Bb * Tt;   // 8192 rows per direction
constexpr int M2 = 2 * M1;    // 16384 rows (fwd + bwd)
}

// ---------------- weight convert + transpose: dst[n*K+k] = bf16(src[k*N+n])
__global__ void k_wT(const float* __restrict__ src, bf16_t* __restrict__ dst,
                     int K, int N, int total) {
  int i = blockIdx.x * 256 + threadIdx.x;
  if (i >= total) return;
  int n = i / K, k = i - n * K;
  dst[i] = (bf16_t)src[(long)k * N + n];
}

// ---------------- row LayerNorm over C=512 (256 threads, 2 elems each)
__global__ void __launch_bounds__(256)
k_ln(const float* __restrict__ src, const float* __restrict__ w,
     const float* __restrict__ b, float* __restrict__ dst, float eps) {
  long row = blockIdx.x;
  const float* xr = src + row * Cc;
  int t = threadIdx.x;
  float a = xr[t], c = xr[t + 256];
  __shared__ float rs[256], rq[256];
  rs[t] = a + c; rq[t] = a * a + c * c;
  __syncthreads();
  for (int s = 128; s > 0; s >>= 1) {
    if (t < s) { rs[t] += rs[t + s]; rq[t] += rq[t + s]; }
    __syncthreads();
  }
  float mu  = rs[0] * (1.f / Cc);
  float var = rq[0] * (1.f / Cc) - mu * mu;
  float inv = rsqrtf(var + eps);
  float* dr = dst + row * Cc;
  dr[t]       = (a - mu) * inv * w[t]       + b[t];
  dr[t + 256] = (c - mu) * inv * w[t + 256] + b[t + 256];
}

// ---------------- bidirectional time-shift diffs: fwd uses x[t-1], bwd x[t+1]
__global__ void k_diff(const float* __restrict__ xn, float* __restrict__ diff,
                       int total /* M1*C */) {
  long i = (long)blockIdx.x * 256 + threadIdx.x;
  if (i >= total) return;
  long row = i / Cc; int t = (int)(row % Tt);
  float cur = xn[i];
  float prv = (t > 0)      ? xn[i - Cc] : 0.f;
  float nxt = (t < Tt - 1) ? xn[i + Cc] : 0.f;
  diff[i]                = prv - cur;   // forward direction
  diff[(long)total + i]  = nxt - cur;   // backward direction
}

__global__ void k_fdiff(const float* __restrict__ xn, float* __restrict__ diff,
                        int total) {
  long i = (long)blockIdx.x * 256 + threadIdx.x;
  if (i >= total) return;
  long row = i / Cc; int t = (int)(row % Tt);
  float cur = xn[i];
  float prv = (t > 0) ? xn[i - Cc] : 0.f;
  diff[i] = prv - cur;
}

// ---------------- WMMA GEMM: C[M,N] = A[M,K] * B[K,N], B given transposed bf16
// MODE 0: A f32; MODE 1: A = xn[row%rowmod] + diff[row]*coef (fused mix);
// MODE 2: A already bf16.
// Block: 256 threads (8 waves), 64x64 macro tile, K-step 32.
// Wave wid: wm=wid&3 -> 16 M-rows, wn=wid>>2 -> 32 N-cols (2 wmma per step).
template <int MODE>
__global__ void __launch_bounds__(256)
k_gemm(const void* __restrict__ Aptr, const float* __restrict__ diffp,
       const float* __restrict__ coef, int rowmod,
       const bf16_t* __restrict__ BT, float* __restrict__ Cout,
       int M, int K, int N) {
  constexpr int AST = 40;  // 80B pitch: 16B aligned, conflict-free mod 64 banks
  constexpr int BST = 40;
  __shared__ bf16_t As[64 * AST];
  __shared__ bf16_t Bs[64 * BST];

  const int bm = blockIdx.y, bn = blockIdx.x;
  const int tid = threadIdx.x;
  const int wid = tid >> 5, lane = tid & 31;
  const int wm = wid & 3, wn = wid >> 2;
  const int lr = lane & 15, hi = lane >> 4;
  const int lrow = tid >> 2;            // 0..63
  const int lcol = (tid & 3) * 8;       // 0,8,16,24

  v8f acc0 = {}; v8f acc1 = {};

  for (int k0 = 0; k0 < K; k0 += 32) {
    // ---- stage A tile (64 rows x 32 k) as bf16
    {
      long grow = (long)bm * 64 + lrow;
      if (MODE == 2) {
        const bf16_t* A = (const bf16_t*)Aptr;
        *(v8bf*)(As + lrow * AST + lcol) =
            *(const v8bf*)(A + grow * K + k0 + lcol);
      } else {
        const float* A = (const float*)Aptr;
        v8bf tv;
        if (MODE == 1) {
          long arow = grow % rowmod;
          const float* xb = A + arow * (long)K + k0 + lcol;
          const float* db = diffp + grow * (long)K + k0 + lcol;
          #pragma unroll
          for (int e = 0; e < 8; ++e)
            tv[e] = (bf16_t)(xb[e] + db[e] * coef[k0 + lcol + e]);
        } else {
          const float* xb = A + grow * (long)K + k0 + lcol;
          #pragma unroll
          for (int e = 0; e < 8; ++e) tv[e] = (bf16_t)xb[e];
        }
        *(v8bf*)(As + lrow * AST + lcol) = tv;
      }
    }
    // ---- stage B tile (64 n-rows x 32 k) from transposed bf16 weights
    {
      long grow = (long)bn * 64 + lrow;
      *(v8bf*)(Bs + lrow * BST + lcol) =
          *(const v8bf*)(BT + grow * K + k0 + lcol);
      if (k0 + 32 < K)  // global_prefetch_b8 for next weight tile
        __builtin_prefetch(BT + grow * K + k0 + 32 + lcol, 0, 3);
    }
    __syncthreads();

    // ---- fragments in contiguous k order (k = 16*hi + s), same for A and B
    union { v16bf v; v8bf h[2]; } af, b0, b1;
    const int ab = hi * 16;      // A: lane<16 -> K 0..15; else K 16..31
    af.h[0] = *(const v8bf*)(As + (wm * 16 + lr) * AST + ab);
    af.h[1] = *(const v8bf*)(As + (wm * 16 + lr) * AST + ab + 8);
    const int kb = hi * 16;      // B: lane<16 -> K 0..15; else K 16..31 (N=lane&15)
    b0.h[0] = *(const v8bf*)(Bs + (wn * 32 + lr) * BST + kb);
    b0.h[1] = *(const v8bf*)(Bs + (wn * 32 + lr) * BST + kb + 8);
    b1.h[0] = *(const v8bf*)(Bs + (wn * 32 + 16 + lr) * BST + kb);
    b1.h[1] = *(const v8bf*)(Bs + (wn * 32 + 16 + lr) * BST + kb + 8);

    acc0 = __builtin_amdgcn_wmma_f32_16x16x32_bf16(
        false, af.v, false, b0.v, (short)0, acc0, false, false);
    acc1 = __builtin_amdgcn_wmma_f32_16x16x32_bf16(
        false, af.v, false, b1.v, (short)0, acc1, false, false);
    __syncthreads();
  }

  // ---- store: vgpr j -> M = j + 8*hi, N = lane&15
  long m0 = (long)bm * 64 + wm * 16 + hi * 8;
  long n0 = (long)bn * 64 + wn * 32 + lr;
  #pragma unroll
  for (int j = 0; j < 8; ++j) {
    Cout[(m0 + j) * N + n0]      = acc0[j];
    Cout[(m0 + j) * N + n0 + 16] = acc1[j];
  }
}

// ---------------- elementwise converters
__global__ void k_tanh_bf16(const float* s, bf16_t* d, long n) {
  long i = (long)blockIdx.x * 256 + threadIdx.x;
  if (i < n) d[i] = (bf16_t)tanhf(s[i]);
}
__global__ void k_cvt_bf16(const float* s, bf16_t* d, long n) {
  long i = (long)blockIdx.x * 256 + threadIdx.x;
  if (i < n) d[i] = (bf16_t)s[i];
}
__global__ void k_sig_bf16(const float* s, bf16_t* d, long n) {
  long i = (long)blockIdx.x * 256 + threadIdx.x;
  if (i < n) d[i] = (bf16_t)(1.f / (1.f + expf(-s[i])));
}
__global__ void k_relu2_bf16(const float* s, bf16_t* d, long n) {
  long i = (long)blockIdx.x * 256 + threadIdx.x;
  if (i < n) { float v = fmaxf(s[i], 0.f); d[i] = (bf16_t)(v * v); }
}

// ---------------- scan prep: decay, a, kk-normalize, k adjust
__global__ void __launch_bounds__(512)
k_scan_prep(float* __restrict__ kbuf, const float* __restrict__ lw,
            const float* __restrict__ la, const float* __restrict__ w0,
            const float* __restrict__ a0, const float* __restrict__ k_k,
            const float* __restrict__ k_a, float* __restrict__ dec,
            float* __restrict__ aab, float* __restrict__ bbb) {
  long m = blockIdx.x; int c = threadIdx.x;
  int h = c >> 6, n = c & 63;
  long idx = m * Cc + c;
  float kv = kbuf[idx];
  float kk = kv * k_k[c];
  __shared__ float red[512];
  red[c] = kk * kk; __syncthreads();
  for (int s = 32; s > 0; s >>= 1) {
    if (n < s) red[c] += red[c + s];
    __syncthreads();
  }
  float nrm = fmaxf(sqrtf(red[h << 6]), 1e-12f);
  float kkn = kk / nrm;
  float u  = w0[c] + lw[idx];
  float sp = log1pf(expf(-u));                 // softplus(-u)
  float d  = expf(-expf(-sp - 0.5f));          // decay in (0,1)
  float av = 1.f / (1.f + expf(-(a0[c] + la[idx])));
  kbuf[idx] = kv * (1.f + (av - 1.f) * k_a[c]);
  dec[idx] = d; aab[idx] = -kkn; bbb[idx] = kkn * av;
}

// ---------------- sequential RWKV-7 scan: 1 block per (dir,b,h), 64 threads,
// thread i owns state row i (64 f32 in VGPRs).
__global__ void __launch_bounds__(64)
k_scan(const float* __restrict__ r, const float* __restrict__ w,
       const float* __restrict__ k, const float* __restrict__ v,
       const float* __restrict__ aa, const float* __restrict__ bb,
       float* __restrict__ out) {
  int bid = blockIdx.x;
  int h = bid & 7, b = (bid >> 3) & 3, dir = bid >> 5;
  int i = threadIdx.x;
  float s[64];
  #pragma unroll
  for (int j = 0; j < 64; ++j) s[j] = 0.f;
  __shared__ float Lr[64], Lw[64], Lk[64], Lv[64], La[64], Lb[64];
  long base = ((long)dir * Bb + b) * Tt;
  for (int ts = 0; ts < Tt; ++ts) {
    int tt = dir ? (Tt - 1 - ts) : ts;
    long row = (base + tt) * Cc + h * Nn;
    Lr[i] = r[row + i]; Lw[i] = w[row + i]; Lk[i] = k[row + i];
    Lv[i] = v[row + i]; La[i] = aa[row + i]; Lb[i] = bb[row + i];
    __syncthreads();
    float sa = 0.f;
    #pragma unroll
    for (int j = 0; j < 64; ++j) sa += s[j] * La[j];
    float vi = Lv[i], o = 0.f;
    #pragma unroll
    for (int j = 0; j < 64; ++j) {
      float ns = s[j] * Lw[j] + sa * Lb[j] + vi * Lk[j];
      s[j] = ns; o += ns * Lr[j];
    }
    out[row + i] = o;
    __syncthreads();
  }
}

// ---------------- GroupNorm(H) + bonus + gate -> bf16 for Wo GEMM
__global__ void __launch_bounds__(512)
k_gn(const float* __restrict__ outs, const float* __restrict__ r,
     const float* __restrict__ k, const float* __restrict__ v,
     const float* __restrict__ g, const float* __restrict__ gn_w,
     const float* __restrict__ gn_b, const float* __restrict__ r_k,
     bf16_t* __restrict__ gated) {
  long m = blockIdx.x; int c = threadIdx.x;
  int h = c >> 6, n = c & 63;
  long idx = m * Cc + c;
  float o  = outs[idx];
  float rb = r[idx] * k[idx] * r_k[c];
  __shared__ float rA[512], rB[512], rC[512];
  rA[c] = o; rB[c] = o * o; rC[c] = rb;
  __syncthreads();
  for (int s = 32; s > 0; s >>= 1) {
    if (n < s) { rA[c] += rA[c + s]; rB[c] += rB[c + s]; rC[c] += rC[c + s]; }
    __syncthreads();
  }
  float mu  = rA[h << 6] * (1.f / 64.f);
  float var = rB[h << 6] * (1.f / 64.f) - mu * mu;
  float gno = (o - mu) * rsqrtf(var + 64e-5f) * gn_w[c] + gn_b[c];
  float res = (gno + rC[h << 6] * v[idx]) * g[idx];
  gated[idx] = (bf16_t)res;
}

// ---------------- residual merges
__global__ void k_residual(const float* xln, const float* att, float* x2,
                           long n) {  // x2 = xln + att_fwd + att_bwd
  long i = (long)blockIdx.x * 256 + threadIdx.x;
  if (i < n) x2[i] = xln[i] + att[i] + att[n + i];
}
__global__ void k_final(const float* x2, const float* ffn, const float* v,
                        float* out, long n) {
  long i = (long)blockIdx.x * 256 + threadIdx.x;
  if (i < n) {
    out[i]     = x2[i] + ffn[i];        // x output
    out[n + i] = v[i] + v[n + i];       // v_out = v_fwd + v_bwd
  }
}

// ---------------------------------------------------------------------------
extern "C" void kernel_launch(void* const* d_in, const int* in_sizes, int n_in,
                              void* d_out, int out_size, void* d_ws,
                              size_t ws_size, hipStream_t stream) {
  (void)in_sizes; (void)n_in; (void)out_size; (void)ws_size;
  // setup_inputs order: x, v_first, then params dict order
  const float* x_in = (const float*)d_in[0];
  const float* pre_w = (const float*)d_in[2];
  const float* pre_b = (const float*)d_in[3];
  const float* an_w = (const float*)d_in[4];
  const float* an_b = (const float*)d_in[5];
  const float* fn_w = (const float*)d_in[6];
  const float* fn_b = (const float*)d_in[7];
  const float* x_r = (const float*)d_in[8];
  const float* x_w = (const float*)d_in[9];
  const float* x_k = (const float*)d_in[10];
  const float* x_v = (const float*)d_in[11];
  const float* x_a = (const float*)d_in[12];
  const float* x_g = (const float*)d_in[13];
  const float* w0  = (const float*)d_in[14];
  const float* w1  = (const float*)d_in[15];
  const float* w2  = (const float*)d_in[16];
  const float* a0  = (const float*)d_in[17];
  const float* a1  = (const float*)d_in[18];
  const float* a2  = (const float*)d_in[19];
  const float* g1  = (const float*)d_in[20];
  const float* g2  = (const float*)d_in[21];
  const float* k_k = (const float*)d_in[22];
  const float* k_a = (const float*)d_in[23];
  const float* r_k = (const float*)d_in[24];
  const float* Wr  = (const float*)d_in[25];
  const float* Wk  = (const float*)d_in[26];
  const float* Wv  = (const float*)d_in[27];
  const float* Wo  = (const float*)d_in[28];
  const float* gn_w = (const float*)d_in[29];
  const float* gn_b = (const float*)d_in[30];
  const float* fx_k = (const float*)d_in[31];
  const float* Wfk  = (const float*)d_in[32];
  const float* Wfv  = (const float*)d_in[33];
  float* out = (float*)d_out;

  // ---- workspace layout
  char* ws = (char*)d_ws;
  size_t off = 0;
  auto F32 = [&](size_t e) { float* p = (float*)(ws + off);
                             off = (off + e * 4 + 255) & ~(size_t)255; return p; };
  auto BF  = [&](size_t e) { bf16_t* p = (bf16_t*)(ws + off);
                             off = (off + e * 2 + 255) & ~(size_t)255; return p; };

  bf16_t* WrT  = BF((size_t)512 * 512);
  bf16_t* WkT  = BF((size_t)512 * 512);
  bf16_t* WvT  = BF((size_t)512 * 512);
  bf16_t* WoT  = BF((size_t)512 * 512);
  bf16_t* w1T  = BF((size_t)64 * 512);
  bf16_t* w2T  = BF((size_t)512 * 64);
  bf16_t* a1T  = BF((size_t)64 * 512);
  bf16_t* a2T  = BF((size_t)512 * 64);
  bf16_t* g1T  = BF((size_t)128 * 512);
  bf16_t* g2T  = BF((size_t)512 * 128);
  bf16_t* WfkT = BF((size_t)2048 * 512);
  bf16_t* WfvT = BF((size_t)512 * 2048);

  float* xln  = F32((size_t)M1 * Cc);
  float* xn   = F32((size_t)M1 * Cc);
  float* diff = F32((size_t)2 * M1 * Cc);
  float* rb   = F32((size_t)M2 * Cc);
  float* kb   = F32((size_t)M2 * Cc);
  float* vb   = F32((size_t)M2 * Cc);
  float* decb = F32((size_t)M2 * Cc);
  float* aab  = F32((size_t)M2 * Cc);
  float* bbb  = F32((size_t)M2 * Cc);
  float* loraw = F32((size_t)M2 * Cc);
  float* loraa = F32((size_t)M2 * Cc);
  float* tmpS  = F32((size_t)M2 * 128);     // LoRA mids (reused w->a->g)
  bf16_t* tmpSb = BF((size_t)M2 * 128);
  float* gbuf  = F32((size_t)M2 * Cc);
  float* outs  = F32((size_t)M2 * Cc);
  bf16_t* gated = BF((size_t)M2 * Cc);
  float* attout = F32((size_t)M2 * Cc);
  float* x2   = F32((size_t)M1 * Cc);
  float* hbuf = F32((size_t)M1 * FF);
  bf16_t* hbb = BF((size_t)M1 * FF);
  float* ffn  = F32((size_t)M1 * Cc);

  auto ew = [&](long n) { return dim3((unsigned)((n + 255) / 256)); };
  auto wt = [&](const float* s, bf16_t* d, int K, int N) {
    int tot = K * N;
    k_wT<<<ew(tot), 256, 0, stream>>>(s, d, K, N, tot);
  };
  auto gemm_mix = [&](const float* A, const float* dp, const float* cf,
                      const bf16_t* BT, float* C, int M, int K, int N) {
    k_gemm<1><<<dim3(N / 64, M / 64), 256, 0, stream>>>(
        (const void*)A, dp, cf, M1, BT, C, M, K, N);
  };
  auto gemm_bf = [&](const bf16_t* A, const bf16_t* BT, float* C, int M, int K,
                     int N) {
    k_gemm<2><<<dim3(N / 64, M / 64), 256, 0, stream>>>(
        (const void*)A, nullptr, nullptr, M1, BT, C, M, K, N);
  };

  // ---- weights -> transposed bf16
  wt(Wr, WrT, 512, 512);  wt(Wk, WkT, 512, 512);
  wt(Wv, WvT, 512, 512);  wt(Wo, WoT, 512, 512);
  wt(w1, w1T, 512, 64);   wt(w2, w2T, 64, 512);
  wt(a1, a1T, 512, 64);   wt(a2, a2T, 64, 512);
  wt(g1, g1T, 512, 128);  wt(g2, g2T, 128, 512);
  wt(Wfk, WfkT, 512, 2048); wt(Wfv, WfvT, 2048, 512);

  // ---- norms + bidirectional shift diffs
  k_ln<<<M1, 256, 0, stream>>>(x_in, pre_w, pre_b, xln, 1e-5f);
  k_ln<<<M1, 256, 0, stream>>>(xln, an_w, an_b, xn, 1e-5f);
  k_diff<<<ew((long)M1 * Cc), 256, 0, stream>>>(xn, diff, M1 * Cc);

  // ---- main projections (mix fused into A staging), both directions at once
  gemm_mix(xn, diff, x_r, WrT, rb, M2, 512, 512);
  gemm_mix(xn, diff, x_k, WkT, kb, M2, 512, 512);
  gemm_mix(xn, diff, x_v, WvT, vb, M2, 512, 512);

  // ---- LoRA w: tanh(xw@w1)@w2
  gemm_mix(xn, diff, x_w, w1T, tmpS, M2, 512, 64);
  k_tanh_bf16<<<ew((long)M2 * 64), 256, 0, stream>>>(tmpS, tmpSb, (long)M2 * 64);
  gemm_bf(tmpSb, w2T, loraw, M2, 64, 512);
  // ---- LoRA a: (xa@a1)@a2
  gemm_mix(xn, diff, x_a, a1T, tmpS, M2, 512, 64);
  k_cvt_bf16<<<ew((long)M2 * 64), 256, 0, stream>>>(tmpS, tmpSb, (long)M2 * 64);
  gemm_bf(tmpSb, a2T, loraa, M2, 64, 512);
  // ---- LoRA g: sigmoid(xg@g1)@g2
  gemm_mix(xn, diff, x_g, g1T, tmpS, M2, 512, 128);
  k_sig_bf16<<<ew((long)M2 * 128), 256, 0, stream>>>(tmpS, tmpSb, (long)M2 * 128);
  gemm_bf(tmpSb, g2T, gbuf, M2, 128, 512);

  // ---- scan prep + sequential scan (64 independent chains)
  k_scan_prep<<<M2, 512, 0, stream>>>(kb, loraw, loraa, w0, a0, k_k, k_a,
                                      decb, aab, bbb);
  k_scan<<<2 * Bb * 8, 64, 0, stream>>>(rb, decb, kb, vb, aab, bbb, outs);

  // ---- GroupNorm + bonus + gate, then output projection
  k_gn<<<M2, 512, 0, stream>>>(outs, rb, kb, vb, gbuf, gn_w, gn_b, r_k, gated);
  gemm_bf(gated, WoT, attout, M2, 512, 512);

  // ---- residual, FFN
  k_residual<<<ew((long)M1 * Cc), 256, 0, stream>>>(xln, attout, x2,
                                                    (long)M1 * Cc);
  k_ln<<<M1, 256, 0, stream>>>(x2, fn_w, fn_b, xn, 1e-5f);
  k_fdiff<<<ew((long)M1 * Cc), 256, 0, stream>>>(xn, diff, M1 * Cc);
  gemm_mix(xn, diff, fx_k, WfkT, hbuf, M1, 512, 2048);
  k_relu2_bf16<<<ew((long)M1 * FF), 256, 0, stream>>>(hbuf, hbb, (long)M1 * FF);
  gemm_bf(hbb, WfvT, ffn, M1, 2048, 512);

  // ---- outputs: x then v_out, each [B,T,C] f32
  k_final<<<ew((long)M1 * Cc), 256, 0, stream>>>(x2, ffn, vb, out,
                                                 (long)M1 * Cc);
}
